// GAT_23149873725489
// MI455X (gfx1250) — matrix operand order repaired and checked by reading, not compiled, not run on hardware
//
#include <hip/hip_runtime.h>
#include <math.h>

typedef _Float16 v16h __attribute__((ext_vector_type(16)));
typedef _Float16 v8h  __attribute__((ext_vector_type(8)));
typedef _Float16 v4h  __attribute__((ext_vector_type(4)));
typedef float    v8f  __attribute__((ext_vector_type(8)));

#define NEG_SLOPE 0.2f

// ---------------------------------------------------------------------------
// CDNA5 async helpers
// ---------------------------------------------------------------------------
__device__ __forceinline__ void async_b128(unsigned lds_byte_off,
                                           unsigned long long gbase,
                                           int gbyte_off) {
  // GLOBAL_LOAD_ASYNC_TO_LDS_B128 (GVS mode): per-lane 16B global -> LDS copy,
  // tracked by ASYNCcnt.
  asm volatile("global_load_async_to_lds_b128 %0, %1, %2"
               :: "v"(lds_byte_off), "v"(gbyte_off), "s"(gbase)
               : "memory");
}

__device__ __forceinline__ void wait_async0() {
#if __has_builtin(__builtin_amdgcn_s_wait_asynccnt)
  __builtin_amdgcn_s_wait_asynccnt(0);
#else
  asm volatile("s_wait_asynccnt 0x0" ::: "memory");
#endif
}

// ---------------------------------------------------------------------------
// Pre-passes: f32 -> f16 flat convert, and W -> f16 transposed [N,K]
// ---------------------------------------------------------------------------
__global__ void convert_f16(const float* __restrict__ in,
                            _Float16* __restrict__ outh, long long n4) {
  long long t = (long long)blockIdx.x * blockDim.x + threadIdx.x;
  if (t >= n4) return;
  const float4 v = ((const float4*)in)[t];
  v4h h; h.x = (_Float16)v.x; h.y = (_Float16)v.y;
         h.z = (_Float16)v.z; h.w = (_Float16)v.w;
  ((v4h*)outh)[t] = h;
}

__global__ void transpose_w_f16(const float* __restrict__ W,
                                _Float16* __restrict__ Wt, int K, int Nc) {
  int t = blockIdx.x * blockDim.x + threadIdx.x;
  if (t >= K * Nc) return;
  int k = t / Nc, n = t % Nc;
  Wt[n * K + k] = (_Float16)W[t];
}

// ---------------------------------------------------------------------------
// WMMA GEMM, async double-buffered staging:
//   out[M,NDIM](f32) = Ah[M,KDIM](f16) x Wt[NDIM,KDIM](f16, pre-transposed)
// 4 waves/block, each wave owns a 16 x NDIM tile; K chunked by 64.
// ---------------------------------------------------------------------------
template <int KDIM, int NDIM>
__global__ __launch_bounds__(128)
void gat_gemm_wmma_async(const _Float16* __restrict__ Ah,
                         const _Float16* __restrict__ Wt,
                         float* __restrict__ out, int M) {
  constexpr int LDH = 72;            // halves per LDS row (144B, 16B-aligned)
  constexpr int ACH = 64 * LDH;      // halves per A buffer
  constexpr int BCH = NDIM * LDH;    // halves per B buffer
  __shared__ __align__(16) _Float16 sA[2 * ACH];
  __shared__ __align__(16) _Float16 sB[2 * BCH];

  const int tid    = threadIdx.x;
  const int lane   = tid & 31;
  const int wv     = tid >> 5;
  const int mblock = blockIdx.x * 64;
  const int lrow   = lane & 15;
  const int hi     = (lane >> 4) & 1;

  const unsigned long long aBase = (unsigned long long)Ah;
  const unsigned long long bBase = (unsigned long long)Wt;
  const unsigned sAoff = (unsigned)(unsigned long long)(&sA[0]); // LDS aperture: low 32 = LDS offset
  const unsigned sBoff = (unsigned)(unsigned long long)(&sB[0]);

  // stage one 64-row x 64-half A chunk into buffer `buf` (512 x b128 / 128 thr)
  auto stageA = [&](int buf, int k0) {
#pragma unroll
    for (int i = 0; i < 4; ++i) {
      int idx = tid + i * 128;
      int r   = idx >> 3;            // row 0..63
      int c8  = idx & 7;             // 8-half chunk 0..7
      int grow = mblock + r; if (grow >= M) grow = M - 1;
      unsigned lds = sAoff + (unsigned)((buf * ACH + r * LDH + c8 * 8) * 2);
      int goff = (int)(((long long)grow * KDIM + k0 + c8 * 8) * 2);
      async_b128(lds, aBase, goff);
    }
  };
  // stage NDIM x 64-half B chunk (Wt rows) into buffer `buf`
  auto stageB = [&](int buf, int k0) {
#pragma unroll
    for (int i = 0; i < (NDIM * 8) / 128; ++i) {
      int idx = tid + i * 128;
      int n   = idx >> 3;
      int c8  = idx & 7;
      unsigned lds = sBoff + (unsigned)((buf * BCH + n * LDH + c8 * 8) * 2);
      int goff = (n * KDIM + k0 + c8 * 8) * 2;
      async_b128(lds, bBase, goff);
    }
  };

  v8f acc[NDIM / 16];
#pragma unroll
  for (int i = 0; i < NDIM / 16; ++i) acc[i] = (v8f){0, 0, 0, 0, 0, 0, 0, 0};

  constexpr int NCH = KDIM / 64;
  stageA(0, 0);
  stageB(0, 0);
  for (int c = 0; c < NCH; ++c) {
    const int buf = c & 1;
    wait_async0();        // my async copies for `buf` done
    __syncthreads();      // everyone's done -> buf readable, buf^1 reusable
    if (c + 1 < NCH) {    // prefetch next chunk while computing this one
      stageA(buf ^ 1, (c + 1) * 64);
      stageB(buf ^ 1, (c + 1) * 64);
    }
#pragma unroll
    for (int ks = 0; ks < 2; ++ks) {
      // A fragment: row lrow, K = {b..b+7} U {b+16..b+23}, b = hi*8
      const _Float16* pa =
          &sA[buf * ACH + (wv * 16 + lrow) * LDH + ks * 32 + hi * 8];
      v8h a_lo = *(const v8h*)pa;
      v8h a_hi = *(const v8h*)(pa + 16);
      v16h af = __builtin_shufflevector(a_lo, a_hi, 0, 1, 2, 3, 4, 5, 6, 7, 8,
                                        9, 10, 11, 12, 13, 14, 15);
#pragma unroll
      for (int ct = 0; ct < NDIM / 16; ++ct) {
        // B fragment: col lrow, K = {b..b+7} U {b+16..b+23}, b = hi*8
        const _Float16* pb =
            &sB[buf * BCH + (ct * 16 + lrow) * LDH + ks * 32 + hi * 8];
        v8h b_lo = *(const v8h*)pb;
        v8h b_hi = *(const v8h*)(pb + 16);
        v16h bf = __builtin_shufflevector(b_lo, b_hi, 0, 1, 2, 3, 4, 5, 6, 7,
                                          8, 9, 10, 11, 12, 13, 14, 15);
        acc[ct] = __builtin_amdgcn_wmma_f32_16x16x32_f16(
            false, af, false, bf, (short)0, acc[ct], false, false);
      }
    }
    __syncthreads();      // all reads of buf done before it is restaged
  }

  // store D: lane holds column (ct*16 + lrow), rows hi*8 + r  (64B coalesced)
#pragma unroll
  for (int ct = 0; ct < NDIM / 16; ++ct) {
    int col = ct * 16 + lrow;
#pragma unroll
    for (int r = 0; r < 8; ++r) {
      int row = mblock + wv * 16 + hi * 8 + r;
      if (row < M) out[(size_t)row * NDIM + col] = acc[ct][r];
    }
  }
}

// ---------------------------------------------------------------------------
// Elementwise / edge kernels
// ---------------------------------------------------------------------------
__global__ void fill_zero(float* __restrict__ p, long long n) {
  long long t = (long long)blockIdx.x * blockDim.x + threadIdx.x;
  if (t < n) p[t] = 0.f;
}

__global__ void alpha_init(const float* __restrict__ h,
                           const float* __restrict__ a_src,
                           const float* __restrict__ a_dst,
                           float* __restrict__ asrc, float* __restrict__ adst,
                           float* __restrict__ m, float* __restrict__ s,
                           int n_nodes, int H, int C) {
  int n = blockIdx.x * blockDim.x + threadIdx.x;
  if (n >= n_nodes) return;
  const float* hr = h + (size_t)n * H * C;
  for (int hd = 0; hd < H; ++hd) {
    float s1 = 0.f, s2 = 0.f;
    for (int c = 0; c < C; ++c) {
      float v = hr[hd * C + c];
      s1 += v * a_src[hd * C + c];
      s2 += v * a_dst[hd * C + c];
    }
    asrc[n * H + hd] = s1;
    adst[n * H + hd] = s2;
    m[n * H + hd]    = -__builtin_inff();
    s[n * H + hd]    = 0.f;
  }
}

__device__ __forceinline__ float leaky(float v) {
  return v > 0.f ? v : NEG_SLOPE * v;
}

// sign-aware float atomic max via int/uint ordering (init must be -inf)
__device__ __forceinline__ void atomicMaxF(float* addr, float val) {
  if (val >= 0.f)
    atomicMax((int*)addr, __float_as_int(val));
  else
    atomicMin((unsigned int*)addr, __float_as_uint(val));
}

__global__ void edge_max(const int* __restrict__ src, const int* __restrict__ dst,
                         const float* __restrict__ asrc,
                         const float* __restrict__ adst,
                         float* __restrict__ m, int E, int H) {
  int e = blockIdx.x * blockDim.x + threadIdx.x;
  if (e >= E) return;
  int sj = src[e], di = dst[e];
  for (int hd = 0; hd < H; ++hd) {
    float v = leaky(asrc[sj * H + hd] + adst[di * H + hd]);
    atomicMaxF(&m[di * H + hd], v);
  }
}

__global__ void edge_expsum(const int* __restrict__ src, const int* __restrict__ dst,
                            const float* __restrict__ asrc,
                            const float* __restrict__ adst,
                            const float* __restrict__ m, float* __restrict__ s,
                            int E, int H) {
  int e = blockIdx.x * blockDim.x + threadIdx.x;
  if (e >= E) return;
  int sj = src[e], di = dst[e];
  for (int hd = 0; hd < H; ++hd) {
    float v = leaky(asrc[sj * H + hd] + adst[di * H + hd]);
    atomicAdd(&s[di * H + hd], __expf(v - m[di * H + hd]));
  }
}

// one thread per (edge, feature-column): coalesced gather + atomic scatter-add
__global__ void edge_msg(const int* __restrict__ src, const int* __restrict__ dst,
                         const float* __restrict__ asrc,
                         const float* __restrict__ adst,
                         const float* __restrict__ m, const float* __restrict__ s,
                         const float* __restrict__ hfeat, float* __restrict__ agg,
                         long long total, int H, int C) {
  long long t = (long long)blockIdx.x * blockDim.x + threadIdx.x;
  if (t >= total) return;
  int HC  = H * C;
  int e   = (int)(t / HC);
  int col = (int)(t % HC);
  int hd  = col / C;
  int sj = src[e], di = dst[e];
  float v = leaky(asrc[sj * H + hd] + adst[di * H + hd]);
  float w = __expf(v - m[di * H + hd]) / s[di * H + hd];
  atomicAdd(&agg[(size_t)di * HC + col], hfeat[(size_t)sj * HC + col] * w);
}

__global__ void elu_bias(float* __restrict__ agg, const float* __restrict__ b,
                         long long total, int NC) {
  long long t = (long long)blockIdx.x * blockDim.x + threadIdx.x;
  if (t >= total) return;
  float v = agg[t] + b[t % NC];
  agg[t] = v > 0.f ? v : (__expf(v) - 1.f);
}

__global__ void final_logsoftmax(const float* __restrict__ agg,
                                 const float* __restrict__ b,
                                 float* __restrict__ out, int n_nodes) {
  int n = blockIdx.x * blockDim.x + threadIdx.x;
  if (n >= n_nodes) return;
  float vals[32];
  float mx = -__builtin_inff();
  for (int c = 0; c < 32; ++c) {
    float v = agg[(size_t)n * 32 + c] + b[c];
    vals[c] = v;
    mx = v > mx ? v : mx;
  }
  float sum = 0.f;
  for (int c = 0; c < 32; ++c) sum += __expf(vals[c] - mx);
  float lse = mx + __logf(sum);
  for (int c = 0; c < 32; ++c) out[(size_t)n * 32 + c] = vals[c] - lse;
}

// ---------------------------------------------------------------------------
extern "C" void kernel_launch(void* const* d_in, const int* in_sizes, int n_in,
                              void* d_out, int out_size, void* d_ws, size_t ws_size,
                              hipStream_t stream) {
  const float* x      = (const float*)d_in[0];
  const int*   eidx   = (const int*)d_in[1];
  const float* W1     = (const float*)d_in[2];
  const float* a_src1 = (const float*)d_in[3];
  const float* a_dst1 = (const float*)d_in[4];
  const float* b1     = (const float*)d_in[5];
  const float* W2     = (const float*)d_in[6];
  const float* a_src2 = (const float*)d_in[7];
  const float* a_dst2 = (const float*)d_in[8];
  const float* b2     = (const float*)d_in[9];
  float*       out    = (float*)d_out;

  const int Nn = in_sizes[0] / 512;
  const int E  = in_sizes[1] / 2;
  const int* src = eidx;       // edge_index row 0
  const int* dst = eidx + E;   // edge_index row 1

  // ---- workspace carve-up: f32 region, then f16 region ----
  float* p   = (float*)d_ws;
  float* h1  = p; p += (size_t)Nn * 64;
  float* as1 = p; p += (size_t)Nn * 8;
  float* ad1 = p; p += (size_t)Nn * 8;
  float* m1  = p; p += (size_t)Nn * 8;
  float* s1  = p; p += (size_t)Nn * 8;
  float* agg1= p; p += (size_t)Nn * 64;   // becomes act1 after ELU
  float* h2  = p; p += (size_t)Nn * 32;
  float* as2 = p; p += (size_t)Nn;
  float* ad2 = p; p += (size_t)Nn;
  float* m2  = p; p += (size_t)Nn;
  float* s2  = p; p += (size_t)Nn;
  float* agg2= p; p += (size_t)Nn * 32;
  _Float16* hp   = (_Float16*)p;
  _Float16* xh   = hp; hp += (size_t)Nn * 512;   // x in f16
  _Float16* a1h  = hp; hp += (size_t)Nn * 64;    // act1 in f16
  _Float16* w1t  = hp; hp += (size_t)512 * 64;   // W1^T f16 [64,512]
  _Float16* w2t  = hp; hp += (size_t)64 * 32;    // W2^T f16 [32,64]

  const int TB = 256;
  // ---------------- layer 1 ----------------
  {
    long long n4 = (long long)Nn * 512 / 4;
    convert_f16<<<(unsigned)((n4 + TB - 1) / TB), TB, 0, stream>>>(x, xh, n4);
  }
  transpose_w_f16<<<(512 * 64 + TB - 1) / TB, TB, 0, stream>>>(W1, w1t, 512, 64);
  gat_gemm_wmma_async<512, 64><<<(Nn + 63) / 64, 128, 0, stream>>>(xh, w1t, h1, Nn);
  alpha_init<<<(Nn + TB - 1) / TB, TB, 0, stream>>>(h1, a_src1, a_dst1, as1, ad1,
                                                    m1, s1, Nn, 8, 8);
  {
    long long z = (long long)Nn * 64;
    fill_zero<<<(unsigned)((z + TB - 1) / TB), TB, 0, stream>>>(agg1, z);
  }
  edge_max<<<(E + TB - 1) / TB, TB, 0, stream>>>(src, dst, as1, ad1, m1, E, 8);
  edge_expsum<<<(E + TB - 1) / TB, TB, 0, stream>>>(src, dst, as1, ad1, m1, s1, E, 8);
  {
    long long tot = (long long)E * 64;
    edge_msg<<<(unsigned)((tot + TB - 1) / TB), TB, 0, stream>>>(
        src, dst, as1, ad1, m1, s1, h1, agg1, tot, 8, 8);
  }
  {
    long long z = (long long)Nn * 64;
    elu_bias<<<(unsigned)((z + TB - 1) / TB), TB, 0, stream>>>(agg1, b1, z, 64);
  }

  // ---------------- layer 2 ----------------
  {
    long long n4 = (long long)Nn * 64 / 4;
    convert_f16<<<(unsigned)((n4 + TB - 1) / TB), TB, 0, stream>>>(agg1, a1h, n4);
  }
  transpose_w_f16<<<(64 * 32 + TB - 1) / TB, TB, 0, stream>>>(W2, w2t, 64, 32);
  gat_gemm_wmma_async<64, 32><<<(Nn + 63) / 64, 128, 0, stream>>>(a1h, w2t, h2, Nn);
  alpha_init<<<(Nn + TB - 1) / TB, TB, 0, stream>>>(h2, a_src2, a_dst2, as2, ad2,
                                                    m2, s2, Nn, 1, 32);
  {
    long long z = (long long)Nn * 32;
    fill_zero<<<(unsigned)((z + TB - 1) / TB), TB, 0, stream>>>(agg2, z);
  }
  edge_max<<<(E + TB - 1) / TB, TB, 0, stream>>>(src, dst, as2, ad2, m2, E, 1);
  edge_expsum<<<(E + TB - 1) / TB, TB, 0, stream>>>(src, dst, as2, ad2, m2, s2, E, 1);
  {
    long long tot = (long long)E * 32;
    edge_msg<<<(unsigned)((tot + TB - 1) / TB), TB, 0, stream>>>(
        src, dst, as2, ad2, m2, s2, h2, agg2, tot, 1, 32);
  }
  final_logsoftmax<<<(Nn + TB - 1) / TB, TB, 0, stream>>>(agg2, b2, out, Nn);
}
